// Head_11510512353363
// MI455X (gfx1250) — matrix operand order repaired and checked by reading, not compiled, not run on hardware
//
#include <hip/hip_runtime.h>

typedef __attribute__((ext_vector_type(16))) _Float16 v16h;
typedef __attribute__((ext_vector_type(8)))  _Float16 v8h;
typedef __attribute__((ext_vector_type(8)))  float    v8f;
typedef __attribute__((ext_vector_type(4)))  float    v4f;
typedef __attribute__((ext_vector_type(4)))  int      v4i;

#define HB 8
#define HT 2048
#define HD 1024
#define HH 64

__device__ __forceinline__ v8f wmma_f16(v16h a, v16h b, v8f c) {
  // D = A(16x32 f16) * B(32x16 f16) + C(16x16 f32)
  return __builtin_amdgcn_wmma_f32_16x16x32_f16(false, a, false, b, (short)0, c,
                                                false, false);
}

// ---- CDNA5 async global->LDS copy (ASYNCcnt-tracked), with sync fallback ----
__device__ __forceinline__ void async_copy16(const _Float16* src, _Float16* ldsdst) {
#if __has_builtin(__builtin_amdgcn_global_load_async_to_lds_b128)
  __builtin_amdgcn_global_load_async_to_lds_b128(
      (v4i __attribute__((address_space(1)))*)(const void*)src,
      (v4i __attribute__((address_space(3)))*)(void*)ldsdst, 0, 0);
#else
  *(v8h*)ldsdst = *(const v8h*)src;
#endif
}
__device__ __forceinline__ void wait_async() {
#if __has_builtin(__builtin_amdgcn_global_load_async_to_lds_b128)
  asm volatile("s_wait_asynccnt 0x0" ::: "memory");
#endif
}

// A fragment (16x32 f16): lane holds row M=lane%16.
// halves 0..7  = K (kb + (lane>>4)*8) .. +7
// halves 8..15 = K (kb + (lane>>4)*8 + 16) .. +23
__device__ __forceinline__ v16h load_a_f16(const _Float16* row, int kb, int lane) {
  int off = kb + ((lane >> 4) << 3);
  v8h lo = *(const v8h*)(row + off);
  v8h hi = *(const v8h*)(row + off + 16);
  v16h a;
#pragma unroll
  for (int i = 0; i < 8; ++i) { a[i] = lo[i]; a[i + 8] = hi[i]; }
  return a;
}

// B fragment (32x16 f16): lane holds column N=lane%16.
// halves 0..7  = K (kb + (lane>>4)*8) .. +7
// halves 8..15 = K (kb + (lane>>4)*8 + 16) .. +23
__device__ __forceinline__ v16h load_b_f16(const _Float16* row, int kb, int lane) {
  int off = kb + ((lane >> 4) << 3);
  v8h lo = *(const v8h*)(row + off);
  v8h hi = *(const v8h*)(row + off + 16);
  v16h b;
#pragma unroll
  for (int i = 0; i < 8; ++i) { b[i] = lo[i]; b[i + 8] = hi[i]; }
  return b;
}

// A fragment built from f32 memory with on-the-fly f16 conversion.
__device__ __forceinline__ v16h load_a_f32(const float* row, int kb, int lane) {
  int off = kb + ((lane >> 4) << 3);
  v4f a0 = *(const v4f*)(row + off);
  v4f a1 = *(const v4f*)(row + off + 4);
  v4f b0 = *(const v4f*)(row + off + 16);
  v4f b1 = *(const v4f*)(row + off + 20);
  v16h a;
#pragma unroll
  for (int i = 0; i < 4; ++i) {
    a[i]      = (_Float16)a0[i];
    a[i + 4]  = (_Float16)a1[i];
    a[i + 8]  = (_Float16)b0[i];
    a[i + 12] = (_Float16)b1[i];
  }
  return a;
}

__device__ __forceinline__ float redmax16(float v) {
#pragma unroll
  for (int off = 8; off >= 1; off >>= 1) v = fmaxf(v, __shfl_xor(v, off, 16));
  return v;
}
__device__ __forceinline__ float redadd16(float v) {
#pragma unroll
  for (int off = 8; off >= 1; off >>= 1) v += __shfl_xor(v, off, 16);
  return v;
}

// ---------------------------------------------------------------------------
// Kernel 0: one-time W transpose+convert: Wt[3][64][1024] f16 (m: 0=K,1=Q,2=V)
// ---------------------------------------------------------------------------
__global__ __launch_bounds__(256) void prep_wt(
    const float* __restrict__ Wk, const float* __restrict__ Wq,
    const float* __restrict__ Wv, _Float16* __restrict__ Wt)
{
  int idx = blockIdx.x * 256 + threadIdx.x;   // 0 .. 196607
  int m = idx >> 16;
  int r = idx & 65535;
  int h = r & 63;
  int k = r >> 6;
  const float* W = (m == 0) ? Wk : ((m == 1) ? Wq : Wv);
  Wt[((size_t)m * HH + h) * HD + k] = (_Float16)W[(size_t)k * HH + h];
}

// ---------------------------------------------------------------------------
// Kernel 1: K/Q/V projections. 4 waves/block, each wave owns 16 rows of x.
// W tiles double-buffered in LDS via async global->LDS b128 copies.
// Writes gK,gQ row-major f16 [B*T,64] and gVt f16 [B,64,T].
// ---------------------------------------------------------------------------
__global__ __launch_bounds__(128) void proj_qkv(
    const float* __restrict__ x,
    const _Float16* __restrict__ Wt,
    const float* __restrict__ bk, const float* __restrict__ bq,
    const float* __restrict__ bv,
    _Float16* __restrict__ gK, _Float16* __restrict__ gQ,
    _Float16* __restrict__ gVt)
{
  __shared__ __align__(16) _Float16 sW[2][3][64][56];  // 112B rows: 16B-aligned
  const int tid  = threadIdx.x;
  const int wave = tid >> 5, lane = tid & 31;
  const int lrow = lane & 15, lhi = lane >> 4;
  const int rowbase = blockIdx.x * 64;
  const int wrow = rowbase + wave * 16 + lrow;

  // stage W[m][0..63][kb..kb+31] f16 into buffer `buf` (768 x 16B chunks)
  auto stage_w = [&](int kb2, int buf) {
#pragma unroll
    for (int i = 0; i < 6; ++i) {
      int c  = tid + i * 128;       // 0..767
      int m  = c >> 8;
      int h  = (c & 255) >> 2;
      int cp = (c & 3) << 3;
      async_copy16(Wt + ((size_t)m * HH + h) * HD + kb2 + cp,
                   &sW[buf][m][h][cp]);
    }
  };

  v8f acc[3][4];
#pragma unroll
  for (int m = 0; m < 3; ++m)
#pragma unroll
    for (int n = 0; n < 4; ++n)
#pragma unroll
      for (int r = 0; r < 8; ++r) acc[m][n][r] = 0.0f;

  const float* xrow = x + (size_t)wrow * HD;

  stage_w(0, 0);
  wait_async();
  __syncthreads();

  for (int it = 0; it < HD / 32; ++it) {
    const int kb = it * 32;
    const int cur = it & 1;
    if (it + 1 < HD / 32) stage_w(kb + 32, cur ^ 1);

    v16h a = load_a_f32(xrow, kb, lane);

    // 12-fragment WMMA batch, B-fragment loads software-pipelined 2 deep so
    // each WMMA overlaps the next fragment's ds_load_b128 pair.
    v16h bcur = load_b_f16(&sW[cur][0][lrow][0], 0, lane);        // f=0 (m0,n0)
    v16h bnxt = load_b_f16(&sW[cur][0][16 + lrow][0], 0, lane);   // f=1 (m0,n1)
#pragma unroll
    for (int f = 0; f < 12; ++f) {
      const int m = f >> 2, n = f & 3;
      acc[m][n] = wmma_f16(a, bcur, acc[m][n]);
      bcur = bnxt;
      if (f + 2 < 12) {
        const int f2 = f + 2, m2 = f2 >> 2, n2 = f2 & 3;
        bnxt = load_b_f16(&sW[cur][m2][n2 * 16 + lrow][0], 0, lane);
      }
    }

    wait_async();
    __syncthreads();
  }

  const int mrow = lhi << 3;
#pragma unroll
  for (int n = 0; n < 4; ++n) {
    int col = n * 16 + lrow;
    float bK = bk[col], bQ = bq[col], bV = bv[col];
#pragma unroll
    for (int r = 0; r < 8; ++r) {
      int row = rowbase + wave * 16 + r + mrow;
      gK[(size_t)row * HH + col] = (_Float16)(acc[0][n][r] + bK);
      gQ[(size_t)row * HH + col] = (_Float16)(acc[1][n][r] + bQ);
      int bb = row >> 11, t = row & (HT - 1);
      gVt[((size_t)bb * HH + col) * HT + t] = (_Float16)(acc[2][n][r] + bV);
    }
  }
}

// ---------------------------------------------------------------------------
// Kernel 2: causal flash attention. Block = 4 waves = 64 queries; 64-key
// blocks double-buffered in LDS via async global->LDS copies; online softmax
// in f32; S and PV via WMMA.
// ---------------------------------------------------------------------------
__global__ __launch_bounds__(128) void attn(
    const _Float16* __restrict__ gQ, const _Float16* __restrict__ gK,
    const _Float16* __restrict__ gVt, float* __restrict__ out)
{
  __shared__ __align__(16) _Float16 sK[2][64][72];    // keys row-major [key][h]
  __shared__ __align__(16) _Float16 sV[2][64][72];    // V transposed [h][key]
  __shared__ __align__(16) _Float16 sP[4][16][72];    // per-wave P staging

  const int tid  = threadIdx.x;
  const int wave = tid >> 5, lane = tid & 31;
  const int lrow = lane & 15, lhi = lane >> 4;
  const int mrow = lhi << 3;
  const int qbase = blockIdx.x * 64;
  const int b = blockIdx.y;
  const float NEG_INF = -__builtin_inff();

  // stage one 64-key K tile + V tile into buffer `buf` (8 x 16B chunks/thread)
  auto stage_kv = [&](int kb2, int buf) {
    const int kbase = kb2 << 6;
#pragma unroll
    for (int i = 0; i < 4; ++i) {
      int c   = tid + i * 128;       // 0..511
      int row = c >> 3;
      int cp  = (c & 7) << 3;
      async_copy16(gK + (size_t)(b * HT + kbase + row) * HH + cp,
                   &sK[buf][row][cp]);
      async_copy16(gVt + ((size_t)b * HH + row) * HT + kbase + cp,
                   &sV[buf][row][cp]);
    }
  };

  const size_t rowQ = (size_t)(b * HT + qbase + wave * 16 + lrow) * HH;
  v16h aq0 = load_a_f16(gQ + rowQ, 0, lane);
  v16h aq1 = load_a_f16(gQ + rowQ, 32, lane);

  float mstat[8], lstat[8];
  v8f o[4];
#pragma unroll
  for (int r = 0; r < 8; ++r) { mstat[r] = NEG_INF; lstat[r] = 0.0f; }
#pragma unroll
  for (int n = 0; n < 4; ++n)
#pragma unroll
    for (int r = 0; r < 8; ++r) o[n][r] = 0.0f;

  const int nkb = (qbase >> 6) + 1;

  stage_kv(0, 0);
  wait_async();
  __syncthreads();

  for (int kb = 0; kb < nkb; ++kb) {
    const int kbase = kb << 6;
    const int cur = kb & 1;
    if (kb + 1 < nkb) stage_kv(kb + 1, cur ^ 1);   // prefetch next tile

    // --- S = Q K^T (16x64 per wave), loads pipelined 2 deep ---
    v8f s[4];
#pragma unroll
    for (int n = 0; n < 4; ++n)
#pragma unroll
      for (int r = 0; r < 8; ++r) s[n][r] = 0.0f;
    {
      v16h bcur = load_b_f16(&sK[cur][lrow][0], 0, lane);    // f=0: n0,k0..31
      v16h bnxt = load_b_f16(&sK[cur][lrow][0], 32, lane);   // f=1: n0,k32..63
#pragma unroll
      for (int f = 0; f < 8; ++f) {
        const int n = f >> 1;
        s[n] = wmma_f16((f & 1) ? aq1 : aq0, bcur, s[n]);
        bcur = bnxt;
        if (f + 2 < 8) {
          const int f2 = f + 2, n2 = f2 >> 1, kk = (f2 & 1) << 5;
          bnxt = load_b_f16(&sK[cur][n2 * 16 + lrow][0], kk, lane);
        }
      }
    }

    // --- scale + causal mask (C layout: row = r+mrow, col = lane%16) ---
#pragma unroll
    for (int n = 0; n < 4; ++n)
#pragma unroll
      for (int r = 0; r < 8; ++r) {
        int qi = qbase + wave * 16 + r + mrow;
        int ki = kbase + n * 16 + lrow;
        float v = s[n][r] * 0.125f;     // 1/sqrt(64)
        s[n][r] = (ki <= qi) ? v : NEG_INF;
      }

    // --- online softmax update ---
#pragma unroll
    for (int r = 0; r < 8; ++r) {
      float lm = fmaxf(fmaxf(s[0][r], s[1][r]), fmaxf(s[2][r], s[3][r]));
      lm = redmax16(lm);
      float mn = fmaxf(mstat[r], lm);
      float alpha = __expf(mstat[r] - mn);
      mstat[r] = mn;
      float ps = 0.0f;
#pragma unroll
      for (int n = 0; n < 4; ++n) {
        float p = __expf(s[n][r] - mn);
        s[n][r] = p;
        ps += p;
      }
      ps = redadd16(ps);
      lstat[r] = lstat[r] * alpha + ps;
#pragma unroll
      for (int n = 0; n < 4; ++n) o[n][r] *= alpha;
    }

    // --- P: C layout -> LDS -> A layout ---
#pragma unroll
    for (int n = 0; n < 4; ++n)
#pragma unroll
      for (int r = 0; r < 8; ++r)
        sP[wave][r + mrow][n * 16 + lrow] = (_Float16)s[n][r];
    __syncthreads();

    // --- O += P V, loads pipelined 2 deep ---
    {
      v16h ap0 = load_a_f16(&sP[wave][lrow][0], 0, lane);
      v16h ap1 = load_a_f16(&sP[wave][lrow][0], 32, lane);
      v16h bcur = load_b_f16(&sV[cur][lrow][0], 0, lane);    // f=0: n0,k0..31
      v16h bnxt = load_b_f16(&sV[cur][lrow][0], 32, lane);   // f=1: n0,k32..63
#pragma unroll
      for (int f = 0; f < 8; ++f) {
        const int n = f >> 1;
        o[n] = wmma_f16((f & 1) ? ap1 : ap0, bcur, o[n]);
        bcur = bnxt;
        if (f + 2 < 8) {
          const int f2 = f + 2, n2 = f2 >> 1, kk = (f2 & 1) << 5;
          bnxt = load_b_f16(&sV[cur][n2 * 16 + lrow][0], kk, lane);
        }
      }
    }

    wait_async();      // next tile landed in the other buffer
    __syncthreads();   // everyone done with sP / sV[cur] before next iter
  }

  // --- normalize + store ---
#pragma unroll
  for (int r = 0; r < 8; ++r) {
    float inv = 1.0f / lstat[r];
    int t = qbase + wave * 16 + r + mrow;
#pragma unroll
    for (int n = 0; n < 4; ++n) {
      int col = n * 16 + lrow;
      out[((size_t)b * HT + t) * HH + col] = o[n][r] * inv;
    }
  }
}

// ---------------------------------------------------------------------------
extern "C" void kernel_launch(void* const* d_in, const int* in_sizes, int n_in,
                              void* d_out, int out_size, void* d_ws, size_t ws_size,
                              hipStream_t stream) {
  const float* x  = (const float*)d_in[0];
  const float* Wk = (const float*)d_in[1];
  const float* bk = (const float*)d_in[2];
  const float* Wq = (const float*)d_in[3];
  const float* bq = (const float*)d_in[4];
  const float* Wv = (const float*)d_in[5];
  const float* bv = (const float*)d_in[6];
  float* out = (float*)d_out;

  const size_t NROW = (size_t)HB * HT;          // 16384
  _Float16* gK  = (_Float16*)d_ws;              // [B*T,64]
  _Float16* gQ  = gK + NROW * HH;               // [B*T,64]
  _Float16* gVt = gQ + NROW * HH;               // [B,64,T]
  _Float16* Wt  = gVt + NROW * HH;              // [3,64,1024]
  // total workspace: 3 * 2MB + 384KB ~= 6.4 MB

  prep_wt<<<dim3(3 * HD * HH / 256), 256, 0, stream>>>(Wk, Wq, Wv, Wt);
  proj_qkv<<<dim3((int)(NROW / 64)), 128, 0, stream>>>(x, Wt, bk, bq, bv,
                                                       gK, gQ, gVt);
  attn<<<dim3(HT / 64, HB), 128, 0, stream>>>(gQ, gK, gVt, out);
}
